// GCN_Encoder_2491081031962
// MI455X (gfx1250) — matrix operand order repaired and checked by reading, not compiled, not run on hardware
//
#include <hip/hip_runtime.h>
#include <hip/hip_bf16.h>
#include <stdint.h>

typedef __attribute__((ext_vector_type(16))) _Float16 v16h;
typedef __attribute__((ext_vector_type(8)))  _Float16 v8h;
typedef __attribute__((ext_vector_type(8)))  float    v8f;

#define FEAT 128
#define NEG_SLOPE 0.01f

// ---------------------------------------------------------------------------
// degree / normalization
// ---------------------------------------------------------------------------
__global__ void k_init_deg(uint32_t* deg, int n) {
    int i = blockIdx.x * blockDim.x + threadIdx.x;
    if (i < n) deg[i] = 1u;                       // self-loop contributes 1
}

__global__ void k_accum_deg(const long long* __restrict__ dst, uint32_t* deg, int e) {
    int i = blockIdx.x * blockDim.x + threadIdx.x;
    if (i < e) atomicAdd(&deg[(int)dst[i]], 1u);
}

__global__ void k_deg_inv_sqrt(const uint32_t* __restrict__ deg, float* dis, int n) {
    int i = blockIdx.x * blockDim.x + threadIdx.x;
    if (i < n) dis[i] = rsqrtf((float)deg[i]);
}

// ---------------------------------------------------------------------------
// f32 -> f16 conversion (for WMMA operands)
// ---------------------------------------------------------------------------
__global__ void k_f32_to_f16(const float* __restrict__ in, _Float16* __restrict__ out, int n) {
    int i = blockIdx.x * blockDim.x + threadIdx.x;
    if (i < n) out[i] = (_Float16)in[i];
}

// ---------------------------------------------------------------------------
// H = Xh @ Wh^T   (N x 128) = (N x 128) @ (128 x 128)^T, f16 in, f32 out.
// One block = one 16-row M tile; 8 waves = the 8 16-col N tiles; K loop 4x32.
// Requires N % 16 == 0 (holds: 100000 = 6250*16); EXEC all-ones for WMMA.
// ---------------------------------------------------------------------------
__global__ void __launch_bounds__(256)
k_wmma_gemm(const _Float16* __restrict__ Xh,   // [N,128] row-major
            const _Float16* __restrict__ Wh,   // [128,128] row-major (W, we use W^T)
            float* __restrict__ H) {           // [N,128] row-major
    const int lane  = threadIdx.x & 31;
    const int wave  = threadIdx.x >> 5;        // 0..7 -> N tile
    const int mbase = blockIdx.x << 4;
    const int nbase = wave << 4;
    const int hsel  = lane >> 4;               // lane half (ISA A/B layout)
    const int lidx  = lane & 15;

    // A: row (mbase+lidx), K chunks [8*hsel,8*hsel+8) and +16 within each 32-K step
    const _Float16* arow = Xh + (size_t)(mbase + lidx) * FEAT + (hsel << 3);
    // B[k][n] = W[n][k]: lane holds column n = nbase+lidx, K chunks [8*hsel,8*hsel+8) and +16
    const _Float16* brow = Wh + (size_t)(nbase + lidx) * FEAT + (hsel << 3);

    v8f acc = {};
#pragma unroll
    for (int kk = 0; kk < 4; ++kk) {
        union { v16h v; v8h h[2]; } a;
        a.h[0] = *(const v8h*)(arow + kk * 32);
        a.h[1] = *(const v8h*)(arow + kk * 32 + 16);
        union { v16h v; v8h h[2]; } bu;
        bu.h[0] = *(const v8h*)(brow + kk * 32);
        bu.h[1] = *(const v8h*)(brow + kk * 32 + 16);
        v16h b = bu.v;
        acc = __builtin_amdgcn_wmma_f32_16x16x32_f16(
            /*neg_a=*/false, a.v, /*neg_b=*/false, b,
            /*c_mod=*/(short)0, acc, /*reuse_a=*/false, /*reuse_b=*/false);
    }

    // D: VGPR v -> row mbase + v + 8*hsel, col nbase + lidx
    float* out = H + (size_t)(mbase + (hsel << 3)) * FEAT + nbase + lidx;
#pragma unroll
    for (int v = 0; v < 8; ++v)
        out[(size_t)v * FEAT] = acc[v];
}

// ---------------------------------------------------------------------------
// agg[i,f] = H[i,f] * (dis[i]^2) + b[f]   (self-loop term + bias; full init)
// ---------------------------------------------------------------------------
__global__ void k_init_agg(const float* __restrict__ H, const float* __restrict__ dis,
                           const float* __restrict__ b, float* __restrict__ agg, int n) {
    int i = blockIdx.x * blockDim.x + threadIdx.x;
    int total = n * FEAT;
    if (i < total) {
        int node = i >> 7;
        int f    = i & (FEAT - 1);
        float sw = dis[node] * dis[node];
        agg[i] = H[i] * sw + b[f];
    }
}

// ---------------------------------------------------------------------------
// scatter: agg[dst] += (dis[src]*dis[dst]) * H[src]; one wave32 per edge,
// float4 per lane (32*4 = 128 feats), native f32 atomics (L2-resident agg).
// ---------------------------------------------------------------------------
__global__ void __launch_bounds__(256)
k_scatter(const long long* __restrict__ src, const long long* __restrict__ dst,
          const float* __restrict__ dis, const float* __restrict__ H,
          float* __restrict__ agg, int e) {
    int edge = (int)((blockIdx.x * blockDim.x + threadIdx.x) >> 5);
    int lane = threadIdx.x & 31;
    if (edge >= e) return;                     // uniform across the wave
    int s = (int)src[edge];
    int d = (int)dst[edge];
    float norm = dis[s] * dis[d];
    float4 hv = ((const float4*)(H + (size_t)s * FEAT))[lane];
    float* arow = agg + (size_t)d * FEAT + lane * 4;
    unsafeAtomicAdd(arow + 0, norm * hv.x);
    unsafeAtomicAdd(arow + 1, norm * hv.y);
    unsafeAtomicAdd(arow + 2, norm * hv.z);
    unsafeAtomicAdd(arow + 3, norm * hv.w);
}

// ---------------------------------------------------------------------------
// in-place leaky relu
// ---------------------------------------------------------------------------
__global__ void k_leaky(float* __restrict__ x, int n) {
    int i = blockIdx.x * blockDim.x + threadIdx.x;
    if (i < n) {
        float v = x[i];
        x[i] = v > 0.0f ? v : NEG_SLOPE * v;
    }
}

// ---------------------------------------------------------------------------
static inline size_t alignUp(size_t x, size_t a) { return (x + a - 1) & ~(a - 1); }

extern "C" void kernel_launch(void* const* d_in, const int* in_sizes, int n_in,
                              void* d_out, int out_size, void* d_ws, size_t ws_size,
                              hipStream_t stream) {
    const float*      x  = (const float*)d_in[0];
    const long long*  ei = (const long long*)d_in[1];   // int64 [2, E]
    const float*      W1 = (const float*)d_in[2];
    const float*      b1 = (const float*)d_in[3];
    const float*      W2 = (const float*)d_in[4];
    const float*      b2 = (const float*)d_in[5];
    float*            out = (float*)d_out;

    const int N  = in_sizes[0] / FEAT;        // 100000
    const int E  = in_sizes[1] / 2;           // 1600000
    const int NF = N * FEAT;                  // 12.8M
    const int WN = FEAT * FEAT;               // 16384
    const long long* src = ei;
    const long long* dst = ei + E;

    // workspace carve-out
    char* ws = (char*)d_ws;
    size_t off = 0;
    uint32_t* deg = (uint32_t*)(ws + off); off = alignUp(off + (size_t)N * 4, 256);
    float*    dis = (float*)(ws + off);    off = alignUp(off + (size_t)N * 4, 256);
    _Float16* xh  = (_Float16*)(ws + off); off = alignUp(off + (size_t)NF * 2, 256);
    float*    Hb  = (float*)(ws + off);    off = alignUp(off + (size_t)NF * 4, 256);
    float*    AGG = (float*)(ws + off);    off = alignUp(off + (size_t)NF * 4, 256);
    _Float16* W1h = (_Float16*)(ws + off); off = alignUp(off + (size_t)WN * 2, 256);
    _Float16* W2h = (_Float16*)(ws + off); off = alignUp(off + (size_t)WN * 2, 256);
    (void)ws_size; (void)n_in; (void)out_size;

    const int T = 256;
    dim3 blk(T);

    // normalization (shared by both layers)
    k_init_deg<<<dim3((N + T - 1) / T), blk, 0, stream>>>(deg, N);
    k_accum_deg<<<dim3((E + T - 1) / T), blk, 0, stream>>>(dst, deg, E);
    k_deg_inv_sqrt<<<dim3((N + T - 1) / T), blk, 0, stream>>>(deg, dis, N);

    // f16 operands
    k_f32_to_f16<<<dim3((WN + T - 1) / T), blk, 0, stream>>>(W1, W1h, WN);
    k_f32_to_f16<<<dim3((WN + T - 1) / T), blk, 0, stream>>>(W2, W2h, WN);
    k_f32_to_f16<<<dim3((NF + T - 1) / T), blk, 0, stream>>>(x, xh, NF);

    dim3 gemm_grid(N / 16);                   // N % 16 == 0
    dim3 scat_grid((E * 32 + T - 1) / T);     // one wave per edge

    // ---- layer 1 ----
    k_wmma_gemm<<<gemm_grid, blk, 0, stream>>>(xh, W1h, Hb);
    k_init_agg<<<dim3((NF + T - 1) / T), blk, 0, stream>>>(Hb, dis, b1, AGG, N);
    k_scatter<<<scat_grid, blk, 0, stream>>>(src, dst, dis, Hb, AGG, E);
    k_leaky<<<dim3((NF + T - 1) / T), blk, 0, stream>>>(AGG, NF);

    // ---- layer 2 ----
    k_f32_to_f16<<<dim3((NF + T - 1) / T), blk, 0, stream>>>(AGG, xh, NF);
    k_wmma_gemm<<<gemm_grid, blk, 0, stream>>>(xh, W2h, Hb);
    k_init_agg<<<dim3((NF + T - 1) / T), blk, 0, stream>>>(Hb, dis, b2, out, N);
    k_scatter<<<scat_grid, blk, 0, stream>>>(src, dst, dis, Hb, out, E);
    k_leaky<<<dim3((NF + T - 1) / T), blk, 0, stream>>>(out, NF);
}
